// ImprovedClusteredCausalAttention_73443940761953
// MI455X (gfx1250) — matrix operand order repaired and checked by reading, not compiled, not run on hardware
//
#include <hip/hip_runtime.h>

#define NB 2
#define LL 2048
#define HH 8
#define EE 64
#define CC 128
#define TOPK 32
#define NEGV -10000000.0f

typedef __attribute__((ext_vector_type(16))) _Float16 v16h;
typedef __attribute__((ext_vector_type(8)))  _Float16 v8h;
typedef __attribute__((ext_vector_type(8)))  float    v8f;

// ---------------------------------------------------------------------------
// 1. bits[nh*L + l] = packed 32-bit sign pattern of [Q(l),1] . planes[b]
// ---------------------------------------------------------------------------
__global__ __launch_bounds__(256) void bits_kernel(const float* __restrict__ queries,
                                                   const float* __restrict__ planes,
                                                   unsigned* __restrict__ bits)
{
    __shared__ float pl[32 * 65];
    for (int i = threadIdx.x; i < 32 * 65; i += 256) pl[i] = planes[i];
    __syncthreads();
    int gid = blockIdx.x * 256 + threadIdx.x;          // (n*H + h)*L + l
    int n = gid >> 14;
    int h = (gid >> 11) & 7;
    int l = gid & (LL - 1);
    const float* q = queries + (((size_t)(n * LL + l) * HH + h) << 6);
    float x[EE];
#pragma unroll
    for (int e = 0; e < EE; ++e) x[e] = q[e];
    unsigned m = 0;
    for (int b = 0; b < 32; ++b) {
        float d = pl[b * 65 + 64];                     // bias term (x[...]=1)
        for (int e = 0; e < EE; ++e) d += x[e] * pl[b * 65 + e];
        if (d > 0.0f) m |= (1u << b);
    }
    bits[gid] = m;
}

// ---------------------------------------------------------------------------
// 2. k-means on packed bits: Hamming distance == popcount(xor). 1 WG per (n,h)
// ---------------------------------------------------------------------------
__global__ __launch_bounds__(256) void cluster_kernel(const unsigned* __restrict__ bits_g,
                                                      unsigned* __restrict__ assign_g)
{
    __shared__ unsigned sb[LL];
    __shared__ unsigned cent[CC];
    __shared__ unsigned cnt[CC];
    __shared__ unsigned sums[CC * 32];
    int nh = blockIdx.x;
    const unsigned* bp = bits_g + (size_t)nh * LL;
    for (int i = threadIdx.x; i < LL; i += 256) sb[i] = bp[i];
    __syncthreads();
    if (threadIdx.x < CC) cent[threadIdx.x] = sb[threadIdx.x * (LL / CC)];
    __syncthreads();

    for (int it = 0; it < 10; ++it) {
        for (int i = threadIdx.x; i < CC; i += 256) cnt[i] = 0;
        for (int i = threadIdx.x; i < CC * 32; i += 256) sums[i] = 0;
        __syncthreads();
        for (int l = threadIdx.x; l < LL; l += 256) {
            unsigned b = sb[l];
            int bestc = 0, bestd = __popc(b ^ cent[0]);
            for (int c = 1; c < CC; ++c) {
                int d = __popc(b ^ cent[c]);
                if (d < bestd) { bestd = d; bestc = c; }
            }
            atomicAdd(&cnt[bestc], 1u);
            unsigned bb = b;
            while (bb) {
                int bit = __ffs(bb) - 1;
                bb &= bb - 1;
                atomicAdd(&sums[bestc * 32 + bit], 1u);
            }
        }
        __syncthreads();
        if (threadIdx.x < CC) {
            unsigned c0 = cnt[threadIdx.x];
            if (c0 > 0) {
                unsigned nc = 0;
                for (int bit = 0; bit < 32; ++bit)
                    if (2u * sums[threadIdx.x * 32 + bit] >= c0) nc |= (1u << bit);
                cent[threadIdx.x] = nc;                // majority vote (>=0.5)
            }
        }
        __syncthreads();
    }
    // final assignment (after 10th centroid update, as in reference)
    for (int l = threadIdx.x; l < LL; l += 256) {
        unsigned b = sb[l];
        int bestc = 0, bestd = __popc(b ^ cent[0]);
        for (int c = 1; c < CC; ++c) {
            int d = __popc(b ^ cent[c]);
            if (d < bestd) { bestd = d; bestc = c; }
        }
        assign_g[(size_t)nh * LL + l] = (unsigned)bestc;
    }
}

// ---------------------------------------------------------------------------
// 3. Qg = per-cluster mean of Q, emitted as f16 row-major [nh][c][e]
// ---------------------------------------------------------------------------
__global__ __launch_bounds__(256) void qg_kernel(const float* __restrict__ queries,
                                                 const unsigned* __restrict__ assign_g,
                                                 _Float16* __restrict__ qgh)
{
    __shared__ float acc[CC * EE];
    __shared__ unsigned cnt[CC];
    int nh = blockIdx.x, n = nh >> 3, h = nh & 7;
    for (int i = threadIdx.x; i < CC * EE; i += 256) acc[i] = 0.0f;
    for (int i = threadIdx.x; i < CC; i += 256) cnt[i] = 0;
    __syncthreads();
    for (int l = threadIdx.x; l < LL; l += 256) {
        unsigned c = assign_g[(size_t)nh * LL + l];
        atomicAdd(&cnt[c], 1u);
        const float* q = queries + (((size_t)(n * LL + l) * HH + h) << 6);
        for (int e = 0; e < EE; ++e) atomicAdd(&acc[c * EE + e], q[e]);
    }
    __syncthreads();
    for (int i = threadIdx.x; i < CC * EE; i += 256) {
        float cN = (float)cnt[i >> 6];
        if (cN < 1.0f) cN = 1.0f;
        qgh[(size_t)nh * CC * EE + i] = (_Float16)(acc[i] / cN);
    }
}

// ---------------------------------------------------------------------------
// 4. K (N,L,H,E) f32  ->  Kh (N,H,S,E) f16 row-major
// ---------------------------------------------------------------------------
__global__ __launch_bounds__(256) void kh_kernel(const float* __restrict__ keys,
                                                 _Float16* __restrict__ kh)
{
    int gid = blockIdx.x * 256 + threadIdx.x;          // ((n*H+h)*S+s)*E+e
    int e = gid & 63;
    int s = (gid >> 6) & (LL - 1);
    int h = (gid >> 17) & 7;
    int n = gid >> 20;
    kh[gid] = (_Float16)keys[(((size_t)(n * LL + s) * HH + h) << 6) + e];
}

// ---------------------------------------------------------------------------
// 5. QKc = Qg @ K^T via v_wmma_f32_16x16x32_f16.
//    Block = 8 waves; each wave owns a 16-column s-tile (B regs hoisted),
//    sweeps 8 c-tiles; K-loop = 2 chunks of 32 -> 2 WMMA per tile.
// ---------------------------------------------------------------------------
__global__ __launch_bounds__(256) void qkc_kernel(const _Float16* __restrict__ qgh,
                                                  const _Float16* __restrict__ kh,
                                                  float* __restrict__ qkc)
{
    int wave = threadIdx.x >> 5;
    int lane = threadIdx.x & 31;
    int sblk = blockIdx.x & 15;
    int nh   = blockIdx.x >> 4;
    int scol = sblk * 128 + wave * 16 + (lane & 15);
    int grp  = lane >> 4;

    // B tile: lane holds column n = lane&15 (= K row scol), 16 contiguous e's
    const _Float16* krow = kh + ((size_t)nh * LL + scol) * EE;
    v16h b0 = *(const v16h*)(krow + grp * 16);         // e = g*16 + 0..15
    v16h b1 = *(const v16h*)(krow + 32 + grp * 16);    // e = 32 + g*16 + 0..15

    const _Float16* qbase = qgh + (size_t)nh * CC * EE;
    float* obase = qkc + (size_t)nh * CC * LL;

    for (int ct = 0; ct < 8; ++ct) {
        // A tile: lane holds row m = lane&15, 16 contiguous e's {g*16+0..15}
        const _Float16* arow = qbase + ((size_t)(ct * 16 + (lane & 15))) * EE;
        v8h a0lo = *(const v8h*)(arow + grp * 16);
        v8h a0hi = *(const v8h*)(arow + grp * 16 + 8);
        v8h a1lo = *(const v8h*)(arow + 32 + grp * 16);
        v8h a1hi = *(const v8h*)(arow + 32 + grp * 16 + 8);
        v16h a0 = __builtin_shufflevector(a0lo, a0hi, 0,1,2,3,4,5,6,7,8,9,10,11,12,13,14,15);
        v16h a1 = __builtin_shufflevector(a1lo, a1hi, 0,1,2,3,4,5,6,7,8,9,10,11,12,13,14,15);
        v8f c = {};
        c = __builtin_amdgcn_wmma_f32_16x16x32_f16(false, a0, false, b0, (short)0, c, false, false);
        c = __builtin_amdgcn_wmma_f32_16x16x32_f16(false, a1, false, b1, (short)0, c, false, false);
#pragma unroll
        for (int r = 0; r < 8; ++r) {
            int m = ct * 16 + r + grp * 8;             // D layout: VGPR r -> row r+g*8
            obase[(size_t)m * LL + scol] = c[r];
        }
    }
}

// ---------------------------------------------------------------------------
// 6. per-row top-32 (iterative argmax over LDS-resident row), 1 WG per row
// ---------------------------------------------------------------------------
__global__ __launch_bounds__(256) void topk_kernel(const float* __restrict__ qkc,
                                                   int* __restrict__ tki)
{
    __shared__ float row[LL];
    __shared__ float rv[256];
    __shared__ int   ri[256];
    int r = blockIdx.x;                                // nh*128 + c
    const float* src = qkc + (size_t)r * LL;
    for (int i = threadIdx.x; i < LL; i += 256) row[i] = src[i];
    __syncthreads();
    for (int k = 0; k < TOPK; ++k) {
        float bv = -3.4e38f; int bi = 0;
        for (int i = threadIdx.x; i < LL; i += 256) {
            float v = row[i];
            if (v > bv) { bv = v; bi = i; }
        }
        rv[threadIdx.x] = bv; ri[threadIdx.x] = bi;
        __syncthreads();
        for (int off = 128; off > 0; off >>= 1) {
            if (threadIdx.x < off) {
                float ov = rv[threadIdx.x + off]; int oi = ri[threadIdx.x + off];
                if (ov > rv[threadIdx.x] ||
                    (ov == rv[threadIdx.x] && oi < ri[threadIdx.x])) {
                    rv[threadIdx.x] = ov; ri[threadIdx.x] = oi;
                }
            }
            __syncthreads();
        }
        if (threadIdx.x == 0) {
            tki[(size_t)r * TOPK + k] = ri[0];
            row[ri[0]] = -3.4e38f;
        }
        __syncthreads();
    }
}

// ---------------------------------------------------------------------------
// 7. final attention: 1 wave per (n,h,l); lane j owns selected key j.
//    f32 throughout (matches reference path). Shuffle softmax, LDS broadcast.
// ---------------------------------------------------------------------------
__global__ __launch_bounds__(256) void attn_kernel(const float* __restrict__ queries,
                                                   const float* __restrict__ keys,
                                                   const float* __restrict__ values,
                                                   const unsigned* __restrict__ assign_g,
                                                   const int* __restrict__ tki,
                                                   float* __restrict__ out)
{
    __shared__ float aw[8][32];
    __shared__ int   ai[8][32];
    int wave = threadIdx.x >> 5, lane = threadIdx.x & 31;
    int gl = blockIdx.x * 8 + wave;                    // (n*H+h)*L + l
    int n = gl >> 14, h = (gl >> 11) & 7, l = gl & (LL - 1);

    unsigned c = assign_g[gl];
    int idx = tki[((size_t)((n * HH + h) * CC + (int)c)) * TOPK + lane];

    const float* q  = queries + (((size_t)(n * LL + l)   * HH + h) << 6);
    const float* kr = keys    + (((size_t)(n * LL + idx) * HH + h) << 6);
    float dot = 0.0f;
#pragma unroll
    for (int e = 0; e < EE; ++e) dot += q[e] * kr[e];

    bool future = idx > l;
    const float temp = 0.125f;                         // 1/sqrt(64)
    float s = future ? (temp * NEGV) : (temp * dot);

    float m = s;
    for (int off = 16; off > 0; off >>= 1) {
        float o = __shfl_xor(m, off, 32);
        m = o > m ? o : m;
    }
    float ex = __expf(s - m);
    float sum = ex;
    for (int off = 16; off > 0; off >>= 1) sum += __shfl_xor(sum, off, 32);
    float a = future ? 0.0f : ex / sum;

    aw[wave][lane] = a; ai[wave][lane] = idx;
    __syncthreads();

    float acc0 = 0.0f, acc1 = 0.0f;
    for (int j = 0; j < TOPK; ++j) {
        float w = aw[wave][j];
        const float* vr = values + (((size_t)(n * LL + ai[wave][j]) * HH + h) << 6);
        acc0 += w * vr[lane];
        acc1 += w * vr[lane + 32];
    }
    float* o = out + (((size_t)(n * LL + l) * HH + h) << 6);
    o[lane]      = acc0;
    o[lane + 32] = acc1;
}

// ---------------------------------------------------------------------------
extern "C" void kernel_launch(void* const* d_in, const int* in_sizes, int n_in,
                              void* d_out, int out_size, void* d_ws, size_t ws_size,
                              hipStream_t stream)
{
    const float* queries = (const float*)d_in[0];
    const float* keys    = (const float*)d_in[1];
    const float* values  = (const float*)d_in[2];
    const float* planes  = (const float*)d_in[3];
    float* out = (float*)d_out;

    char* ws = (char*)d_ws;
    size_t off = 0;
    auto carve = [&](size_t bytes) {
        void* p = ws + off;
        off += (bytes + 255) & ~(size_t)255;
        return p;
    };
    const int NHL = NB * HH * LL;                      // 32768
    unsigned*  bits   = (unsigned*)carve((size_t)NHL * 4);
    unsigned*  assign = (unsigned*)carve((size_t)NHL * 4);
    _Float16*  qgh    = (_Float16*)carve((size_t)NB * HH * CC * EE * 2);
    _Float16*  kh     = (_Float16*)carve((size_t)NB * HH * LL * EE * 2);
    float*     qkc    = (float*)   carve((size_t)NB * HH * CC * LL * 4);
    int*       tki    = (int*)     carve((size_t)NB * HH * CC * TOPK * 4);
    (void)ws_size; (void)in_sizes; (void)n_in; (void)out_size;

    bits_kernel   <<<NHL / 256, 256, 0, stream>>>(queries, planes, bits);
    cluster_kernel<<<NB * HH,   256, 0, stream>>>(bits, assign);
    qg_kernel     <<<NB * HH,   256, 0, stream>>>(queries, assign, qgh);
    kh_kernel     <<<(NB * HH * LL * EE) / 256, 256, 0, stream>>>(keys, kh);
    qkc_kernel    <<<NB * HH * (LL / 128), 256, 0, stream>>>(qgh, kh, qkc);
    topk_kernel   <<<NB * HH * CC, 256, 0, stream>>>(qkc, tki);
    attn_kernel   <<<NHL / 8, 256, 0, stream>>>(queries, keys, values, assign, tki, out);
}
